// MMD_AU_45234595562184
// MI455X (gfx1250) — matrix operand order repaired and checked by reading, not compiled, not run on hardware
//
#include <hip/hip_runtime.h>

typedef __attribute__((ext_vector_type(16))) _Float16 v16h;
typedef __attribute__((ext_vector_type(8)))  _Float16 v8h;
typedef __attribute__((ext_vector_type(8)))  float    v8f;

#define NS     2048
#define TWO_N  4096
#define DF     64
#define NPER   128
#define NK     6
#define HPITCH 40   // padded i-row pitch (halfs): 80B rows -> conflict-free banks

// ---------------------------------------------------------------------------
// Prep: Z = concat(X,Y) -> f16 copy + per-row squared norms (f32)
// ---------------------------------------------------------------------------
__global__ void mmd_prep_z(const float* __restrict__ X, const float* __restrict__ Y,
                           _Float16* __restrict__ Zh, float* __restrict__ sq) {
    int row = blockIdx.x * blockDim.x + threadIdx.x;
    if (row >= TWO_N) return;
    const float* src = (row < NS) ? (X + (size_t)row * DF)
                                  : (Y + (size_t)(row - NS) * DF);
    float s = 0.0f;
#pragma unroll
    for (int d = 0; d < DF; ++d) {
        float v = src[d];
        s += v * v;
        Zh[(size_t)row * DF + d] = (_Float16)v;
    }
    sq[row] = s;
}

// ---------------------------------------------------------------------------
// Prep: S = 2*B - 1 as f16 (exact +-1)
// ---------------------------------------------------------------------------
__global__ void mmd_prep_s(const int* __restrict__ B, _Float16* __restrict__ Sf) {
    int idx = blockIdx.x * blockDim.x + threadIdx.x;
    if (idx >= NPER * NS) return;
    Sf[idx] = (_Float16)(2 * B[idx] - 1);
}

// ---------------------------------------------------------------------------
// Dist GEMM: Dist[r][c] = sqrt(max(sq[r]+sq[c]-2*(Zh_r . Zh_c), 0))
// One 16x16 tile per wave, K=64 via two chained WMMA f16 16x16x32.
// ---------------------------------------------------------------------------
__global__ void __launch_bounds__(256)
mmd_dist(const _Float16* __restrict__ Zh, const float* __restrict__ sq,
         float* __restrict__ Dist) {
    const int lane = threadIdx.x & 31;
    const int wave = threadIdx.x >> 5;
    const int n    = lane & 15;
    const int hi   = lane >> 4;
    const int cbase = blockIdx.x * 16;
    const int rbase = (blockIdx.y * 8 + wave) * 16;

    const int mrow = rbase + n;   // A-matrix row for this lane (M = lane&15)
    const int ncol = cbase + n;   // B-matrix column for this lane

    v8f acc = {};
#pragma unroll
    for (int k0 = 0; k0 < DF; k0 += 32) {
        v16h a, b;
        const _Float16* ap = Zh + (size_t)mrow * DF + k0;
        const _Float16* bp = Zh + (size_t)ncol * DF + k0 + hi * 16;
#pragma unroll
        for (int p = 0; p < 8; ++p) {
            a[p]     = ap[hi * 16 + p];          // K = k0 + hi*16 + p
            a[8 + p] = ap[hi * 16 + 8 + p];      // K = k0 + hi*16 + 8 + p
        }
#pragma unroll
        for (int p = 0; p < 16; ++p) b[p] = bp[p];  // K = k0 + hi*16 + p
        acc = __builtin_amdgcn_wmma_f32_16x16x32_f16(
                  false, a, false, b, (short)0, acc, false, false);
    }

    const float sqc = sq[ncol];
#pragma unroll
    for (int s = 0; s < 8; ++s) {
        int r = rbase + hi * 8 + s;              // C slot s -> M = hi*8 + s
        float d2 = fmaxf(sq[r] + sqc - 2.0f * acc[s], 0.0f);
        // single hardware v_sqrt_f32; f16-GEMM input makes IEEE sqrt pointless
        Dist[(size_t)r * TWO_N + ncol] = __builtin_amdgcn_sqrtf(d2);
    }
}

// ---------------------------------------------------------------------------
// Main fused phase: per j-tile (16 cols), loop i in steps of 32:
//   - compute h_k[32x16] tiles for all 6 kernel fns into LDS (exp from Dist)
//   - 8 waves each do T_k(Mtile) += S(Mtile,K=32) x h_k(32x16)  via WMMA
// Epilogue: U partial sums + quadratic-form partials s^T h s via shfl+atomics.
// ---------------------------------------------------------------------------
__global__ void __launch_bounds__(256)
mmd_main(const float* __restrict__ Dist, const _Float16* __restrict__ Sf,
         const float* __restrict__ bandwidths,
         float* __restrict__ Usum, float* __restrict__ Ubsum) {
    // [kernel][j-col][i-row(padded)] : 6*16*40*2 = 7680 B
    __shared__ __align__(32) _Float16 hlds[NK][16][HPITCH];

    const int tid  = threadIdx.x;
    const int lane = tid & 31;
    const int wave = tid >> 5;
    const int n    = lane & 15;
    const int hi   = lane >> 4;
    const int jb   = blockIdx.x * 16;

    float coef[NK];
#pragma unroll
    for (int k = 0; k < NK; ++k) {
        float bw = bandwidths[k];
        coef[k] = (k < 3) ? 1.0f / (bw * bw)   // gaussian: exp(-d^2/bw^2)
                          : 1.0f / bw;         // laplacian: exp(-d/bw)
    }

    v8f zero = {};
    v8f acc[NK];
    float hsum[NK];
#pragma unroll
    for (int k = 0; k < NK; ++k) { acc[k] = zero; hsum[k] = 0.0f; }

    for (int ibase = 0; ibase < NS; ibase += 32) {
        // Prefetch next i-step's Dist rows (L2-resident; gfx1250 global_prefetch)
        if (ibase + 32 < NS) {
            int an = ibase + 32 + lane;
            __builtin_prefetch(&Dist[(size_t)an * TWO_N + jb], 0, 1);
            __builtin_prefetch(&Dist[(size_t)(NS + an) * TWO_N + NS + jb], 0, 1);
            __builtin_prefetch(&Dist[(size_t)an * TWO_N + NS + jb], 0, 1);
            __builtin_prefetch(&Dist[(size_t)(NS + an) * TWO_N + jb], 0, 1);
        }

        // --- cooperative fill of h tiles (all 6 kernel fns) ---
        // q = tid + 256*e: nn = q&15 (lane-consecutive j-col -> coalesced Dist
        // loads, 64B segments per 16 lanes), ii = q>>4 (i-row).
#pragma unroll
        for (int e = 0; e < 2; ++e) {
            int q  = tid + 256 * e;
            int nn = q & 15;
            int ii = q >> 4;
            int a  = ibase + ii;
            int jc = jb + nn;
            size_t ra  = (size_t)a * TWO_N;
            size_t rya = (size_t)(NS + a) * TWO_N;
            float d1 = Dist[ra  + jc];        // ||X_a - X_j||
            float d2 = Dist[rya + NS + jc];   // ||Y_a - Y_j||
            float d3 = Dist[ra  + NS + jc];   // ||X_a - Y_j||  (K_XY[a,j])
            float d4 = Dist[rya + jc];        // ||X_j - Y_a||  (K_XY[j,a], symmetry)
            float s1 = d1 * d1, s2 = d2 * d2, s3 = d3 * d3, s4 = d4 * d4;
            float mask = (a == jc) ? 0.0f : 1.0f;
#pragma unroll
            for (int k = 0; k < NK; ++k) {
                float c = coef[k];
                float h;
                if (k < 3)
                    h = __expf(-s1 * c) + __expf(-s2 * c)
                      - __expf(-s3 * c) - __expf(-s4 * c);
                else
                    h = __expf(-d1 * c) + __expf(-d2 * c)
                      - __expf(-d3 * c) - __expf(-d4 * c);
                h *= mask;
                hsum[k] += h;
                hlds[k][nn][ii] = (_Float16)h;   // banks 20*nn mod 64: conflict-free
            }
        }
        __syncthreads();

        // --- WMMA: A = S tile (16 perms x 32 i), shared across the 6 kernels ---
        v16h afrag;
        {
            const _Float16* sp = Sf + (size_t)(wave * 16 + n) * NS + ibase;
#pragma unroll
            for (int p = 0; p < 8; ++p) {
                afrag[p]     = sp[hi * 16 + p];
                afrag[8 + p] = sp[hi * 16 + 8 + p];
            }
        }
#pragma unroll
        for (int k = 0; k < NK; ++k) {
            // rows are only 16B aligned (pitch 80B): load two aligned v8h, concat
            const v8h* bp = (const v8h*)&hlds[k][n][hi * 16];
            v8h b0 = bp[0];
            v8h b1 = bp[1];
            v16h bfrag = __builtin_shufflevector(b0, b1,
                0, 1, 2, 3, 4, 5, 6, 7, 8, 9, 10, 11, 12, 13, 14, 15);
            acc[k] = __builtin_amdgcn_wmma_f32_16x16x32_f16(
                         false, afrag, false, bfrag, (short)0, acc[k], false, false);
        }
        __syncthreads();
    }

    // --- U[k] partial: reduce hsum across wave, one atomic per wave ---
#pragma unroll
    for (int k = 0; k < NK; ++k) {
        float v = hsum[k];
        for (int m = 1; m < 32; m <<= 1) v += __shfl_xor(v, m, 32);
        if (lane == 0) atomicAdd(&Usum[k], v);
    }

    // --- U_b partials: acc[k][s] = T[b][j], b = wave*16 + hi*8 + s, j = jb+n ---
#pragma unroll
    for (int k = 0; k < NK; ++k) {
#pragma unroll
        for (int s = 0; s < 8; ++s) {
            int b  = wave * 16 + hi * 8 + s;
            int jc = jb + n;
            float sgn = (float)Sf[(size_t)b * NS + jc];
            float v = acc[k][s] * sgn;
            for (int m = 1; m < 16; m <<= 1) v += __shfl_xor(v, m, 32);
            if (n == 0) atomicAdd(&Ubsum[k * NPER + b], v);
        }
    }
}

// ---------------------------------------------------------------------------
// Finalize: out = [U (6), U_b (6x128)] / (n*(n-1))
// ---------------------------------------------------------------------------
__global__ void mmd_final(const float* __restrict__ Usum,
                          const float* __restrict__ Ubsum,
                          float* __restrict__ out) {
    int t = blockIdx.x * blockDim.x + threadIdx.x;
    const float invden = 1.0f / ((float)NS * (float)(NS - 1));
    if (t < NK) out[t] = Usum[t] * invden;
    else if (t < NK + NK * NPER) out[t] = Ubsum[t - NK] * invden;
}

// ---------------------------------------------------------------------------
// Host-side launcher
// ---------------------------------------------------------------------------
extern "C" void kernel_launch(void* const* d_in, const int* in_sizes, int n_in,
                              void* d_out, int out_size, void* d_ws, size_t ws_size,
                              hipStream_t stream) {
    const float* X  = (const float*)d_in[0];      // [2048,64]
    const float* Y  = (const float*)d_in[1];      // [2048,64]
    const float* bw = (const float*)d_in[2];      // [6]
    const int*   B  = (const int*)d_in[3];        // [128,2048]
    float* out = (float*)d_out;                   // 6 + 768

    char* ws = (char*)d_ws;
    const size_t OFF_DIST  = 0;                                    // 67108864 B
    const size_t OFF_ZH    = OFF_DIST + (size_t)TWO_N * TWO_N * 4; // 524288 B
    const size_t OFF_SF    = OFF_ZH   + (size_t)TWO_N * DF * 2;    // 524288 B
    const size_t OFF_SQ    = OFF_SF   + (size_t)NPER * NS * 2;     // 16384 B
    const size_t OFF_USUM  = OFF_SQ   + (size_t)TWO_N * 4;
    const size_t OFF_UBSUM = OFF_USUM + 32;

    float*    Dist  = (float*)(ws + OFF_DIST);
    _Float16* Zh    = (_Float16*)(ws + OFF_ZH);
    _Float16* Sf    = (_Float16*)(ws + OFF_SF);
    float*    sq    = (float*)(ws + OFF_SQ);
    float*    Usum  = (float*)(ws + OFF_USUM);
    float*    Ubsum = (float*)(ws + OFF_UBSUM);

    // zero accumulators (graph-capture safe memset node)
    hipMemsetAsync(ws + OFF_USUM, 0, 32 + (size_t)NK * NPER * 4, stream);

    mmd_prep_z<<<TWO_N / 256, 256, 0, stream>>>(X, Y, Zh, sq);
    mmd_prep_s<<<(NPER * NS + 255) / 256, 256, 0, stream>>>(B, Sf);

    // 256 c-tiles x 32 r-groups(8 waves each) = full 4096x4096
    mmd_dist<<<dim3(TWO_N / 16, TWO_N / 128), 256, 0, stream>>>(Zh, sq, Dist);

    // one workgroup per 16-column j-tile
    mmd_main<<<NS / 16, 256, 0, stream>>>(Dist, Sf, bw, Usum, Ubsum);

    mmd_final<<<4, 256, 0, stream>>>(Usum, Ubsum, out);
}
